// MoETransceiverVQ_49864570306944
// MI455X (gfx1250) — matrix operand order repaired and checked by reading, not compiled, not run on hardware
//
#include <hip/hip_runtime.h>
#include <hip/hip_bf16.h>
#include <math.h>

// ---------------------------------------------------------------------------
// Types for CDNA5 WMMA (wave32): v_wmma_f32_16x16x32_bf16
// ---------------------------------------------------------------------------
typedef __attribute__((ext_vector_type(16))) __bf16 v16bf;
typedef __attribute__((ext_vector_type(8)))  float  v8f;
typedef unsigned short ushort_t;

union Frag32B { uint4 u[2]; v16bf v; };

__device__ __forceinline__ unsigned short f2bf(float f) {
  union { float f; unsigned int u; } x; x.f = f;
  unsigned int r = x.u + 0x7FFFu + ((x.u >> 16) & 1u);  // round-to-nearest-even
  return (unsigned short)(r >> 16);
}
__device__ __forceinline__ unsigned int pack2(float a, float b) {
  return (unsigned int)f2bf(a) | ((unsigned int)f2bf(b) << 16);
}
__device__ __forceinline__ uint4 cvt8(float4 lo, float4 hi) {
  uint4 r;
  r.x = pack2(lo.x, lo.y); r.y = pack2(lo.z, lo.w);
  r.z = pack2(hi.x, hi.y); r.w = pack2(hi.z, hi.w);
  return r;
}

// Problem constants
#define BB 64
#define NN 1024
#define CC 256
#define KK 1024
#define TOTAL_BITS 10240
#define PAD_BITS 10248
// 1.25 / (B*N*C*C)
#define VQ_SCALE 2.9103830456733704e-10f

// ---------------------------------------------------------------------------
// Kernel 0: per-codeword 0.5*||c||^2 AND one-time f32->bf16 codebook copy
// (codeword-major / K-contiguous == exact WMMA B-frag layout, b128 loads)
// ---------------------------------------------------------------------------
__global__ void k_prep(const float* __restrict__ cb, float* __restrict__ hcbsq,
                       uint4* __restrict__ cb16) {
  int cw = blockIdx.x * 256 + threadIdx.x;   // 0..4095  (r*1024 + k)
  const float* row = cb + (size_t)cw * CC;
  uint4* orow = cb16 + (size_t)cw * 32;      // 256 bf16 = 32 uint4 per codeword
  float s = 0.f;
  #pragma unroll 4
  for (int c = 0; c < CC; c += 8) {
    float4 v0 = *(const float4*)(row + c);
    float4 v1 = *(const float4*)(row + c + 4);
    s += v0.x*v0.x + v0.y*v0.y + v0.z*v0.z + v0.w*v0.w;
    s += v1.x*v1.x + v1.y*v1.y + v1.z*v1.z + v1.w*v1.w;
    orow[c >> 3] = cvt8(v0, v1);
  }
  hcbsq[cw] = 0.5f * s;
}

// ---------------------------------------------------------------------------
// Kernel 1: routing MLP (tiny), softmax, argmax; zero vq accumulator
// ---------------------------------------------------------------------------
__global__ void k_mlp(const float* __restrict__ phi,
                      const float* __restrict__ w0, const float* __restrict__ b0,
                      const float* __restrict__ w1, const float* __restrict__ b1,
                      const float* __restrict__ w2, const float* __restrict__ b2,
                      float* __restrict__ logits_out, float* __restrict__ probs_out,
                      int* __restrict__ mode_out,
                      int* __restrict__ expert_ws, int* __restrict__ phy_ws,
                      float* __restrict__ vq_out) {
  __shared__ float sp[512];
  __shared__ float h1[128];
  __shared__ float h2[128];
  __shared__ float lg[12];
  const int t = threadIdx.x;  // 128 threads
  if (t == 0) *vq_out = 0.0f;
  for (int b = 0; b < BB; ++b) {
    for (int k = t; k < 512; k += 128) sp[k] = phi[b*512 + k];
    __syncthreads();
    float a = b0[t];
    for (int k = 0; k < 512; ++k) a += sp[k] * w0[k*128 + t];
    h1[t] = fmaxf(a, 0.f);
    __syncthreads();
    float c = b1[t];
    for (int k = 0; k < 128; ++k) c += h1[k] * w1[k*128 + t];
    h2[t] = fmaxf(c, 0.f);
    __syncthreads();
    if (t < 12) {
      float l = b2[t];
      for (int k = 0; k < 128; ++k) l += h2[k] * w2[k*12 + t];
      lg[t] = l;
      logits_out[b*12 + t] = l;
    }
    __syncthreads();
    if (t == 0) {
      float mx = lg[0]; int am = 0;
      for (int j = 1; j < 12; ++j) if (lg[j] > mx) { mx = lg[j]; am = j; }
      float e[12]; float den = 0.f;
      for (int j = 0; j < 12; ++j) { e[j] = expf(lg[j] - mx); den += e[j]; }
      float inv = 1.0f / den;
      for (int j = 0; j < 12; ++j) probs_out[b*12 + j] = e[j] * inv;
      mode_out[b] = am;
      expert_ws[b] = am / 3;
      phy_ws[b] = am % 3;
    }
    __syncthreads();
  }
}

// ---------------------------------------------------------------------------
// Kernel 2: VQ nearest-codeword via bf16 WMMA GEMM + running argmax.
// Block = 256 threads = 8 waves = 8 row-tiles (128 rows); each wave sweeps
// all 64 column tiles of its expert's codebook (B stream shared via WGP$).
// Software-pipelined: double-buffered B fragments so loads of tile j+1
// overlap the 8 WMMAs of tile j (partial loadcnt waits instead of 0).
// Branchless argmax update (v_cndmask, no exec-mask branches).
// ---------------------------------------------------------------------------
__device__ __forceinline__ void loadB(Frag32B (&f)[8], float& hq,
                                      const ushort_t* __restrict__ cb16base,
                                      const float* __restrict__ hcb,
                                      int col, int laneHi) {
  const uint4* crow = (const uint4*)(cb16base + (size_t)col * CC) + laneHi;
  #pragma unroll
  for (int kc = 0; kc < 8; ++kc) {
    f[kc].u[0] = crow[kc*4];
    f[kc].u[1] = crow[kc*4 + 2];
  }
  hq = hcb[col];
}

__device__ __forceinline__ void tileMMA(const Frag32B (&af)[8],
                                        const Frag32B (&f)[8],
                                        float hq, int col,
                                        float (&bestv)[8], int (&besti)[8]) {
  v8f acc0 = {}, acc1 = {};
  #pragma unroll
  for (int kc = 0; kc < 8; kc += 2) {
    acc0 = __builtin_amdgcn_wmma_f32_16x16x32_bf16(
        false, af[kc].v,   false, f[kc].v,   (short)0, acc0, false, false);
    acc1 = __builtin_amdgcn_wmma_f32_16x16x32_bf16(
        false, af[kc+1].v, false, f[kc+1].v, (short)0, acc1, false, false);
  }
  #pragma unroll
  for (int v = 0; v < 8; ++v) {
    float sc = (acc0[v] + acc1[v]) - hq;
    bool gt = sc > bestv[v];
    besti[v] = gt ? col : besti[v];
    bestv[v] = gt ? sc  : bestv[v];
  }
}

__global__ __launch_bounds__(256)
void k_vq(const float* __restrict__ H, const ushort_t* __restrict__ cb16,
          const float* __restrict__ cb,
          const float* __restrict__ hcbsq, const int* __restrict__ expert,
          float* __restrict__ zq_out, int* __restrict__ ind_out,
          float* __restrict__ vq_out) {
  const int tid  = threadIdx.x;
  const int lane = tid & 31;
  const int wave = tid >> 5;          // row-tile 0..7
  const int laneHi = lane >> 4;
  const int rowBase = blockIdx.x * 128;
  const int b = rowBase >> 10;        // 8 blocks per batch
  const int r = expert[b];

  // ---- A fragments: 16 rows x 256 K in registers (8 bf16 frags), converted
  // once per block, amortized over 64 column tiles.
  // 16-bit A 16x32 layout: lanes 0-15 -> K {0..7,16..23}; lanes 16-31 -> {8..15,24..31}
  const int grow  = rowBase + wave*16 + (lane & 15);
  const int khalf = laneHi * 8;
  const float* zr = H + (size_t)grow * CC;
  Frag32B af[8];
  #pragma unroll
  for (int kc = 0; kc < 8; ++kc) {
    const int kb = kc*32 + khalf;
    float4 a0 = *(const float4*)(zr + kb);
    float4 a1 = *(const float4*)(zr + kb + 4);
    float4 a2 = *(const float4*)(zr + kb + 16);
    float4 a3 = *(const float4*)(zr + kb + 20);
    af[kc].u[0] = cvt8(a0, a1);
    af[kc].u[1] = cvt8(a2, a3);
  }

  const ushort_t* cb16base = cb16 + (size_t)r * (KK * CC);
  const float*    hcb      = hcbsq + r * KK;
  const int laneCol = lane & 15;

  float bestv[8]; int besti[8];
  #pragma unroll
  for (int v = 0; v < 8; ++v) { bestv[v] = -3.4e38f; besti[v] = 0; }

  // ---- sweep all 64 codeword tiles, double-buffered ----
  Frag32B f0[8], f1[8];
  float hq0, hq1;
  loadB(f0, hq0, cb16base, hcb, laneCol, laneHi);
  #pragma unroll 1
  for (int j = 0; j < 64; j += 2) {
    const int col0 = j*16 + laneCol;
    const int col1 = col0 + 16;
    loadB(f1, hq1, cb16base, hcb, col1, laneHi);          // tile j+1 in flight
    if (j + 2 < 64)
      __builtin_prefetch(cb16base + (size_t)(col0 + 32) * CC, 0, 1);
    tileMMA(af, f0, hq0, col0, bestv, besti);             // consume tile j
    if (j + 2 < 64)
      loadB(f0, hq0, cb16base, hcb, col1 + 16, laneHi);   // tile j+2 in flight
    tileMMA(af, f1, hq1, col1, bestv, besti);             // consume tile j+1
  }

  // ---- per-row reduce: columns live across 16 lanes of each half-wave ----
  #pragma unroll
  for (int v = 0; v < 8; ++v) {
    float bv = bestv[v]; int bi = besti[v];
    #pragma unroll
    for (int off = 1; off < 16; off <<= 1) {
      float ov = __shfl_xor(bv, off, 32);
      int   oi = __shfl_xor(bi, off, 32);
      if (ov > bv || (ov == bv && oi < bi)) { bv = ov; bi = oi; }
    }
    bestv[v] = bv; besti[v] = bi;
  }

  __shared__ int s_fi[128];
  if ((lane & 15) == 0) {
    const int rl = wave*16 + laneHi*8;   // C layout: slot v -> row v (+8 upper half)
    #pragma unroll
    for (int v = 0; v < 8; ++v) s_fi[rl + v] = besti[v];
  }
  __syncthreads();
  if (tid < 128) ind_out[rowBase + tid] = s_fi[tid];

  // ---- write z_q_st rows (f32 codebook gather), accumulate squared error ----
  const int rl = tid >> 1;              // 128 rows, 2 threads per row
  const int cseg = (tid & 1) * 128;
  const int fi = s_fi[rl];
  const float* cw = cb + ((size_t)(r * KK + fi)) * CC + cseg;
  const float* hr = H + (size_t)(rowBase + rl) * CC + cseg;
  float* zo = zq_out + (size_t)(rowBase + rl) * CC + cseg;
  float lsum = 0.f;
  #pragma unroll 4
  for (int c = 0; c < 128; c += 4) {
    float4 q = *(const float4*)(cw + c);
    float4 h = *(const float4*)(hr + c);
    float dx = q.x - h.x, dy = q.y - h.y, dz = q.z - h.z, dw = q.w - h.w;
    float4 o = make_float4(h.x + dx, h.y + dy, h.z + dz, h.w + dw);
    *(float4*)(zo + c) = o;
    lsum += dx*dx + dy*dy + dz*dz + dw*dw;
  }
  __shared__ float s_red[256];
  s_red[tid] = lsum;
  __syncthreads();
  for (int s2 = 128; s2 > 0; s2 >>= 1) {
    if (tid < s2) s_red[tid] += s_red[tid + s2];
    __syncthreads();
  }
  if (tid == 0) atomicAdd(vq_out, s_red[0] * VQ_SCALE);
}

// ---------------------------------------------------------------------------
// Kernel 3: QAM modulate + AWGN + grid demodulate -> received bit bytes
// (deterministic counter-based PCG/Box-Muller noise)
// ---------------------------------------------------------------------------
__device__ __forceinline__ unsigned int pcg(unsigned int x) {
  x = x * 747796405u + 2891336453u;
  unsigned int w = ((x >> ((x >> 28) + 4u)) ^ x) * 277803737u;
  return (w >> 22) ^ w;
}

__global__ void k_chan(const int* __restrict__ ind, const int* __restrict__ phy,
                       const float* __restrict__ noise_var,
                       unsigned char* __restrict__ bh) {
  const int b = blockIdx.y;
  const int s = blockIdx.x * 256 + threadIdx.x;
  const int m = phy[b];
  const int bps = 2 + 2*m;
  const int n_sym = PAD_BITS / bps;     // 5124 / 2562 / 1708
  if (s >= n_sym) return;
  const int half = bps >> 1;
  const int ms = 1 << half;             // 2 / 4 / 8
  const float scl = sqrtf(2.0f * (float)(ms*ms - 1) / 3.0f + 1e-9f);
  const float inv = 1.0f / scl;

  const int p0 = s * bps;
  int sym = 0;
  for (int t = 0; t < bps; ++t) {
    int p = p0 + t;
    int bit = 0;
    if (p < TOTAL_BITS) bit = (ind[b*NN + p/10] >> (9 - p%10)) & 1;
    sym = (sym << 1) | bit;
  }
  const int i = sym >> half, j = sym & (ms - 1);
  float tx_x = (float)(2*i - (ms - 1)) * inv;
  float tx_y = (float)(2*j - (ms - 1)) * inv;

  const float stdv = sqrtf(fmaxf(noise_var[0], 0.f));
  unsigned int st = (unsigned)b * 0x9E3779B9u ^ (unsigned)s * 0x85EBCA6Bu ^
                    (unsigned)m * 0xC2B2AE35u;
  unsigned int r1 = pcg(st), r2 = pcg(st ^ 0x6C8E9CF5u);
  float u1 = (float)(r1 >> 8) * 5.9604645e-08f + 1e-12f;
  float u2 = (float)(r2 >> 8) * 5.9604645e-08f;
  float rad = sqrtf(-2.0f * logf(u1));
  float n0 = rad * cosf(6.28318530718f * u2);
  float n1 = rad * sinf(6.28318530718f * u2);

  float rx_x = tx_x + stdv * n0;
  float rx_y = tx_y + stdv * n1;
  int li = (int)rintf((rx_x * scl + (float)(ms - 1)) * 0.5f);
  int lj = (int)rintf((rx_y * scl + (float)(ms - 1)) * 0.5f);
  li = min(max(li, 0), ms - 1);
  lj = min(max(lj, 0), ms - 1);
  const int ridx = li * ms + lj;

  unsigned char* o = bh + (size_t)b * PAD_BITS + p0;
  for (int t = 0; t < bps; ++t)
    o[t] = (unsigned char)((ridx >> (bps - 1 - t)) & 1);
}

// ---------------------------------------------------------------------------
// Kernel 4: reassemble idx_hat (10 bits) and gather H_hat rows
// ---------------------------------------------------------------------------
__global__ void k_asm(const unsigned char* __restrict__ bh,
                      const int* __restrict__ expert,
                      const float* __restrict__ H, const float* __restrict__ cb,
                      float* __restrict__ hhat, int* __restrict__ ih_out) {
  const int row = blockIdx.x * 4 + (threadIdx.x >> 6);  // 65536 rows
  const int l = threadIdx.x & 63;
  const int b = row >> 10;
  const int n = row & (NN - 1);
  const unsigned char* p = bh + (size_t)b * PAD_BITS + n * 10;
  int ih = 0;
  #pragma unroll
  for (int t = 0; t < 10; ++t) ih = (ih << 1) | (int)p[t];
  if (ih > KK - 1) ih = KK - 1;     // jnp.clip (no-op for 10 bits, kept for parity)
  if (l == 0) ih_out[row] = ih;
  const int r = expert[b];
  const float* cw = cb + ((size_t)(r * KK + ih)) * CC + l * 4;
  const float* hr = H + (size_t)row * CC + l * 4;
  float4 q = *(const float4*)cw;
  float4 h = *(const float4*)hr;
  float4 o = make_float4(h.x + (q.x - h.x), h.y + (q.y - h.y),
                         h.z + (q.z - h.z), h.w + (q.w - h.w));
  *(float4*)(hhat + (size_t)row * CC + l * 4) = o;
}

// ---------------------------------------------------------------------------
extern "C" void kernel_launch(void* const* d_in, const int* in_sizes, int n_in,
                              void* d_out, int out_size, void* d_ws, size_t ws_size,
                              hipStream_t stream) {
  const float* H   = (const float*)d_in[0];
  const float* phi = (const float*)d_in[1];
  const float* w0  = (const float*)d_in[2];
  const float* b0  = (const float*)d_in[3];
  const float* w1  = (const float*)d_in[4];
  const float* b1  = (const float*)d_in[5];
  const float* w2  = (const float*)d_in[6];
  const float* b2  = (const float*)d_in[7];
  const float* cb  = (const float*)d_in[8];
  const float* nv  = (const float*)d_in[9];

  float* out  = (float*)d_out;
  float* hhat = out;                              // 16777216
  float* zq   = out + 16777216;                   // 16777216
  int*   ind  = (int*)(out + 33554432);           // 65536
  int*   ihat = (int*)(out + 33619968);           // 65536
  float* vq   = out + 33685504;                   // 1
  float* lgt  = out + 33685505;                   // 768
  float* prb  = out + 33686273;                   // 768
  int*   mode = (int*)(out + 33687041);           // 64

  char* ws = (char*)d_ws;
  ushort_t* cb16     = (ushort_t*)ws;                        // 4*1024*256*2 = 2 MiB
  float* hcbsq       = (float*)(ws + 2097152);               // 4096 f32
  int*   expert      = (int*)(ws + 2097152 + 16384);         // 64 i32
  int*   phyidx      = (int*)(ws + 2097152 + 16384 + 256);   // 64 i32
  unsigned char* bh  = (unsigned char*)(ws + 2097152 + 16384 + 512); // 64*10248 B

  k_prep<<<dim3(16), dim3(256), 0, stream>>>(cb, hcbsq, (uint4*)cb16);
  k_mlp<<<dim3(1), dim3(128), 0, stream>>>(phi, w0, b0, w1, b1, w2, b2,
                                           lgt, prb, mode, expert, phyidx, vq);
  k_vq<<<dim3(512), dim3(256), 0, stream>>>(H, cb16, cb, hcbsq, expert, zq, ind, vq);
  k_chan<<<dim3(21, 64), dim3(256), 0, stream>>>(ind, phyidx, nv, bh);
  k_asm<<<dim3(16384), dim3(256), 0, stream>>>(bh, expert, H, cb, hhat, ihat);
}
